// GGUFWeight_1571958030524
// MI455X (gfx1250) — compile-verified, matrix operand order checked by reading, not run on hardware
//
#include <hip/hip_runtime.h>

typedef __attribute__((ext_vector_type(16))) __bf16 v16bf;
typedef __attribute__((ext_vector_type(8)))  __bf16 v8bf;
typedef __attribute__((ext_vector_type(4)))  __bf16 v4bf;
typedef __attribute__((ext_vector_type(8)))  float  v8f;
typedef __attribute__((ext_vector_type(4)))  float  v4f;

#define K_IN   4096
#define N_OUT  4096
#define U16_PER_ROW (K_IN / 32 * 9)   /* 1152 u16 = 2304 bytes per weight row */
#define BM 128                         /* CTA rows (tokens)   */
#define BN 256                         /* CTA cols (out feats)*/
#define BK 32                          /* K-step = one q4_0 block */
#define LDSK 40                        /* padded bf16 row stride (80 B, 16B aligned) */

__global__ __launch_bounds__(256) void q4_gemm_wmma(
    const float* __restrict__ x,
    const unsigned short* __restrict__ qw,   // u16 view of qweight
    const float* __restrict__ bias,
    float* __restrict__ out)
{
  // ping-pong LDS: A 2*128*40*2B = 20 KB, B 2*256*40*2B = 40 KB  (60 KB total)
  __shared__ __align__(16) __bf16 sA[2][BM * LDSK];
  __shared__ __align__(16) __bf16 sB[2][BN * LDSK];

  const int tid  = threadIdx.x;
  const int lane = tid & 31;
  const int wid  = tid >> 5;
  const int wm   = wid >> 2;          // 0..1 : 64-row stripe within CTA tile
  const int wn   = wid & 3;           // 0..3 : 64-col stripe

  const int m0 = blockIdx.y * BM;
  const int n0 = blockIdx.x * BN;

  // A staging: 8 float4 per row, 1024 float4 per tile, 4 per thread
  const int acol  = tid & 7;          // float4 column within K-step
  const int arow0 = tid >> 3;         // row base, +32 per i
  // B staging: one weight row (one q4_0 block per K-step) per thread
  const unsigned short* qrow = qw + (size_t)(n0 + tid) * U16_PER_ROW;

  // ---- staged registers (next tile) ----
  v4f            areg[4];
  unsigned short bscale;
  unsigned short bd[8];

  auto load_regs = [&](int kt) {
    #pragma unroll
    for (int i = 0; i < 4; ++i) {
      const float* gp = x + (size_t)(m0 + arow0 + i * 32) * K_IN + kt + acol * 4;
      areg[i] = *(const v4f*)gp;
      if (kt + BK < K_IN)
        __builtin_prefetch((const void*)(gp + BK), 0, 3);   // warm kt+2 tile
    }
    const unsigned short* bp = qrow + (size_t)(kt / 32) * 9;
    bscale = bp[0];
    #pragma unroll
    for (int j = 0; j < 8; ++j) bd[j] = bp[1 + j];
  };

  auto store_lds = [&](int b) {
    // A: fp32 -> bf16
    #pragma unroll
    for (int i = 0; i < 4; ++i) {
      v4bf ab = { (__bf16)areg[i].x, (__bf16)areg[i].y,
                  (__bf16)areg[i].z, (__bf16)areg[i].w };
      *(v4bf*)&sA[b][(arow0 + i * 32) * LDSK + acol * 4] = ab;
    }
    // B: q4_0 dequant -> bf16 (byte j: lo nibble = k j, hi nibble = k j+16)
    union { unsigned short u; _Float16 h; } cv; cv.u = bscale;
    const float s = (float)cv.h;
    __bf16 wv[32];
    #pragma unroll
    for (int j = 0; j < 8; ++j) {
      const int b0 = bd[j] & 0xFF;
      const int b1 = (bd[j] >> 8) & 0xFF;
      wv[2*j]      = (__bf16)(((b0 & 15) - 8) * s);
      wv[2*j + 16] = (__bf16)(((b0 >> 4) - 8) * s);
      wv[2*j + 1]  = (__bf16)(((b1 & 15) - 8) * s);
      wv[2*j + 17] = (__bf16)(((b1 >> 4) - 8) * s);
    }
    __bf16* dst = &sB[b][tid * LDSK];
    #pragma unroll
    for (int c = 0; c < 4; ++c)
      *(v8bf*)(dst + c * 8) = *(v8bf*)&wv[c * 8];
  };

  v8f acc[4][4];
  const v8f vzero = {0.f,0.f,0.f,0.f,0.f,0.f,0.f,0.f};
  #pragma unroll
  for (int i = 0; i < 4; ++i)
    #pragma unroll
    for (int j = 0; j < 4; ++j)
      acc[i][j] = vzero;

  // ---- prologue: stage first tile ----
  load_regs(0);
  store_lds(0);
  __syncthreads();

  int buf = 0;
  for (int kt = 0; kt < K_IN; kt += BK) {
    const bool has_next = (kt + BK) < K_IN;
    if (has_next) load_regs(kt + BK);          // global loads fly during compute

    // ---- compute on current buffer: 4x4 tiles of 16x16, K=32 ----
    // WMMA 16-bit A layout: lanes 0-15 -> K 0-7 & 16-23; lanes 16-31 -> K 8-15 & 24-31
    const int kb = (lane >> 4) * 8;

    auto loadB = [&](int tn) -> v16bf {
      const __bf16* p = &sB[buf][(wn * 64 + tn * 16 + (lane & 15)) * LDSK + kb];
      v8bf lo = *(const v8bf*)p;
      v8bf hi = *(const v8bf*)(p + 16);
      return __builtin_shufflevector(lo, hi, 0,1,2,3,4,5,6,7,8,9,10,11,12,13,14,15);
    };

    v16bf afr[4];
    #pragma unroll
    for (int tm = 0; tm < 4; ++tm) {
      const __bf16* p = &sA[buf][(wm * 64 + tm * 16 + (lane & 15)) * LDSK + kb];
      v8bf lo = *(const v8bf*)p;
      v8bf hi = *(const v8bf*)(p + 16);
      afr[tm] = __builtin_shufflevector(lo, hi, 0,1,2,3,4,5,6,7,8,9,10,11,12,13,14,15);
    }

    // B-fragment software pipeline: keep 3 live, issue loads 2 groups ahead
    v16bf bcur = loadB(0);
    v16bf bnxt = loadB(1);
    #pragma unroll
    for (int tn = 0; tn < 4; ++tn) {
      v16bf bfut = bnxt;
      if (tn < 2) bfut = loadB(tn + 2);        // issued before consuming bcur
      #pragma unroll
      for (int tm = 0; tm < 4; ++tm)
        acc[tm][tn] = __builtin_amdgcn_wmma_f32_16x16x32_bf16(
            false, afr[tm], false, bcur, (short)0, acc[tm][tn], false, false);
      bcur = bnxt;
      bnxt = bfut;
    }

    if (has_next) store_lds(buf ^ 1);          // fill the other buffer
    __syncthreads();
    buf ^= 1;
  }

  // ---- epilogue: bias + store f32 ----
  // C/D layout: VGPR r -> M = r (lanes 0-15) or r+8 (lanes 16-31); N = lane & 15
  #pragma unroll
  for (int tn = 0; tn < 4; ++tn) {
    const int n = n0 + wn * 64 + tn * 16 + (lane & 15);
    const float bv = bias[n];
    #pragma unroll
    for (int tm = 0; tm < 4; ++tm) {
      const int mbase = m0 + wm * 64 + tm * 16 + (lane >> 4) * 8;
      #pragma unroll
      for (int r = 0; r < 8; ++r)
        out[(size_t)(mbase + r) * N_OUT + n] = acc[tm][tn][r] + bv;
    }
  }
}

extern "C" void kernel_launch(void* const* d_in, const int* in_sizes, int n_in,
                              void* d_out, int out_size, void* d_ws, size_t ws_size,
                              hipStream_t stream) {
  const float* x            = (const float*)d_in[0];
  const unsigned short* qw  = (const unsigned short*)d_in[1];  // u16 view of uint8 qweight
  const float* bias         = (const float*)d_in[2];
  float* out                = (float*)d_out;

  const int M = in_sizes[0] / K_IN;          // 8192 tokens
  dim3 grid(N_OUT / BN, M / BM);             // (16, 64)
  q4_gemm_wmma<<<grid, 256, 0, stream>>>(x, qw, bias, out);
}
